// MultiheadLatentAttention_40827959116467
// MI455X (gfx1250) — compile-verified, matrix operand order checked by reading, not run on hardware
//
#include <hip/hip_runtime.h>
#include <math.h>

#define B_   2
#define S_   1024
#define DM_  2048
#define H_   16
#define HD_  128
#define RD_  64
#define DQL_ 384
#define DKVL_ 512

typedef __bf16 bf16_t;
typedef __attribute__((ext_vector_type(16))) __bf16 v16bf;
typedef __attribute__((ext_vector_type(8)))  __bf16 v8bf;
typedef __attribute__((ext_vector_type(8)))  float  v8f;

__device__ inline void store_c(float* p, float v)  { *p = v; }
__device__ inline void store_c(bf16_t* p, float v) { *p = (bf16_t)v; }

// CDNA5 async global->LDS copy (16B per lane), tracked by ASYNCcnt
__device__ inline void async_copy_b128(uint32_t lds_off, const bf16_t* gptr) {
  asm volatile("global_load_async_to_lds_b128 %0, %1, off"
               :: "v"(lds_off), "v"(gptr) : "memory");
}

// ---------------------------------------------------------------------------
// Elementwise converts
// ---------------------------------------------------------------------------
__global__ void k_f32_to_bf16(const float* __restrict__ in, bf16_t* __restrict__ out, int n) {
  int i = blockIdx.x * blockDim.x + threadIdx.x;
  if (i < n) out[i] = (bf16_t)in[i];
}

// out[n*K + k] = bf16(in[k*N + n])   (weight transpose so GEMM B-operand is K-contiguous)
__global__ void k_transpose_bf16(const float* __restrict__ in, bf16_t* __restrict__ out, int K, int N) {
  int i = blockIdx.x * blockDim.x + threadIdx.x;
  if (i < K * N) {
    int n = i / K, k = i % K;
    out[i] = (bf16_t)in[(size_t)k * N + n];
  }
}

// ---------------------------------------------------------------------------
// RMSNorm: one row per block (n = 384 or 512), f32 in -> bf16 out
// ---------------------------------------------------------------------------
__global__ void k_rmsnorm(const float* __restrict__ in, const float* __restrict__ w,
                          bf16_t* __restrict__ out, int n) {
  __shared__ float red[128];
  int row = blockIdx.x;
  const float* r = in + (size_t)row * n;
  float ss = 0.f;
  for (int i = threadIdx.x; i < n; i += 128) { float v = r[i]; ss += v * v; }
  red[threadIdx.x] = ss; __syncthreads();
  for (int off = 64; off > 0; off >>= 1) {
    if (threadIdx.x < off) red[threadIdx.x] += red[threadIdx.x + off];
    __syncthreads();
  }
  float inv = rsqrtf(red[0] / n + 1e-6f);
  bf16_t* o = out + (size_t)row * n;
  for (int i = threadIdx.x; i < n; i += 128) o[i] = (bf16_t)(r[i] * inv * w[i]);
}

// ---------------------------------------------------------------------------
// Pack q/k dense + rope halves into qcat/kcat [b,h,s,192] bf16, and v^T [b,h,d,s]
// ---------------------------------------------------------------------------
__global__ void k_pack(const float* __restrict__ q_raw, const float* __restrict__ k_raw,
                       const float* __restrict__ v_raw,
                       const float* __restrict__ qr_raw, const float* __restrict__ kr_raw,
                       bf16_t* __restrict__ qcat, bf16_t* __restrict__ kcat,
                       bf16_t* __restrict__ vt) {
  int z = blockIdx.x;                 // (b*H + h)*S + s
  int s = z % S_;
  int h = (z / S_) % H_;
  int b = z / (S_ * H_);
  int t = threadIdx.x;                // 0..127
  size_t rowm   = (size_t)(b * S_ + s);
  size_t densei = rowm * (H_ * HD_) + h * HD_ + t;
  size_t cbase  = (size_t)z * (HD_ + RD_);
  qcat[cbase + t] = (bf16_t)q_raw[densei];
  kcat[cbase + t] = (bf16_t)k_raw[densei];
  vt[(size_t)((b * H_ + h) * HD_ + t) * S_ + s] = (bf16_t)v_raw[densei];
  if (t < RD_ / 2) {
    int i = t;
    float freq = __powf(10000.0f, -2.0f * i / RD_);
    float ang = (float)s * freq;
    float sn, cs; __sincosf(ang, &sn, &cs);
    size_t rbase = rowm * (H_ * RD_) + h * RD_ + 2 * i;
    float qe = qr_raw[rbase], qo = qr_raw[rbase + 1];
    float ke = kr_raw[rbase], ko = kr_raw[rbase + 1];
    qcat[cbase + HD_ + 2 * i]     = (bf16_t)(qe * cs - qo * sn);
    qcat[cbase + HD_ + 2 * i + 1] = (bf16_t)(qe * sn + qo * cs);
    kcat[cbase + HD_ + 2 * i]     = (bf16_t)(ke * cs - ko * sn);
    kcat[cbase + HD_ + 2 * i + 1] = (bf16_t)(ke * sn + ko * cs);
  }
}

// ---------------------------------------------------------------------------
// Row softmax over S=1024: in-place f32 (final attn output) + bf16 copy
// ---------------------------------------------------------------------------
__global__ void k_softmax(float* __restrict__ attn, bf16_t* __restrict__ attn_bf) {
  __shared__ float red[256];
  size_t row = blockIdx.x;
  float* p = attn + row * (size_t)S_;
  int t = threadIdx.x;
  float v[4];
  float m = -1e30f;
  #pragma unroll
  for (int j = 0; j < 4; j++) { v[j] = p[t + j * 256]; m = fmaxf(m, v[j]); }
  red[t] = m; __syncthreads();
  for (int off = 128; off > 0; off >>= 1) {
    if (t < off) red[t] = fmaxf(red[t], red[t + off]);
    __syncthreads();
  }
  m = red[0]; __syncthreads();
  float s = 0.f;
  #pragma unroll
  for (int j = 0; j < 4; j++) { v[j] = __expf(v[j] - m); s += v[j]; }
  red[t] = s; __syncthreads();
  for (int off = 128; off > 0; off >>= 1) {
    if (t < off) red[t] += red[t + off];
    __syncthreads();
  }
  float inv = 1.0f / red[0];
  bf16_t* pb = attn_bf + row * (size_t)S_;
  #pragma unroll
  for (int j = 0; j < 4; j++) { float a = v[j] * inv; p[t + j * 256] = a; pb[t + j * 256] = (bf16_t)a; }
}

// [b,h,s,d] -> [b,s,h*HD+d]
__global__ void k_repack_out(const bf16_t* __restrict__ in, bf16_t* __restrict__ out) {
  int i = blockIdx.x * blockDim.x + threadIdx.x;
  if (i >= B_ * H_ * S_ * HD_) return;
  int d = i % HD_;
  int s = (i / HD_) % S_;
  int h = (i / (HD_ * S_)) % H_;
  int b = i / (HD_ * S_ * H_);
  out[((size_t)(b * S_ + s)) * (H_ * HD_) + h * HD_ + d] = in[i];
}

// ---------------------------------------------------------------------------
// Generic batched bf16 GEMM: C[z] = alpha * A[z] (MxK) * Bt[z]^T (Bt is NxK)
// 256 threads = 8 waves; block tile 128x128; each wave 32x64 (8 x 16x16 WMMAs)
// Double-buffered LDS staged with GLOBAL_LOAD_ASYNC_TO_LDS_B128 (ASYNCcnt).
// ---------------------------------------------------------------------------
template <typename OutT>
__global__ __launch_bounds__(256)
void k_gemm_bf16(const bf16_t* __restrict__ A, const bf16_t* __restrict__ Bt,
                 OutT* __restrict__ C,
                 int M, int N, int K, int lda, int ldb, int ldc,
                 long long Az, long long Bz, long long Cz, float alpha) {
  __shared__ bf16_t As[2][128][48];   // 48-halfword stride: pad + 16B alignment
  __shared__ bf16_t Bs[2][128][48];

  const int z = blockIdx.z;
  A  += (size_t)z * Az;
  Bt += (size_t)z * Bz;
  C  += (size_t)z * Cz;

  const int row0 = blockIdx.y * 128;
  const int col0 = blockIdx.x * 128;
  const int tid  = threadIdx.x;
  const int wave = tid >> 5;
  const int lane = tid & 31;
  const int l    = lane & 15;
  const int hi2  = lane >> 4;
  const int wm   = (wave >> 1) * 32;   // 0,32,64,96
  const int wn   = (wave & 1) * 64;    // 0,64

  // staging: 128 rows x 4 16B-chunks per matrix = 512 chunks; 2 per thread
  const int c0 = tid,        r0 = c0 >> 2, cc0 = (c0 & 3) * 8;
  const int c1 = tid + 256,  r1 = c1 >> 2, cc1 = (c1 & 3) * 8;

  auto stage = [&](int buf, int k0) {
    async_copy_b128((uint32_t)(uintptr_t)&As[buf][r0][cc0],
                    A + (size_t)(row0 + r0) * lda + k0 + cc0);
    async_copy_b128((uint32_t)(uintptr_t)&As[buf][r1][cc1],
                    A + (size_t)(row0 + r1) * lda + k0 + cc1);
    async_copy_b128((uint32_t)(uintptr_t)&Bs[buf][r0][cc0],
                    Bt + (size_t)(col0 + r0) * ldb + k0 + cc0);
    async_copy_b128((uint32_t)(uintptr_t)&Bs[buf][r1][cc1],
                    Bt + (size_t)(col0 + r1) * ldb + k0 + cc1);
  };

  v8f acc[2][4];
  #pragma unroll
  for (int mi = 0; mi < 2; mi++)
    #pragma unroll
    for (int ni = 0; ni < 4; ni++)
      #pragma unroll
      for (int r = 0; r < 8; r++) acc[mi][ni][r] = 0.0f;

  stage(0, 0);                         // prologue: fill buffer 0
  int buf = 0;

  for (int k0 = 0; k0 < K; k0 += 32) {
    if (k0 + 32 < K) {
      stage(buf ^ 1, k0 + 32);         // prefetch next K-step into other buffer
      asm volatile("s_wait_asynccnt 4" ::: "memory");  // current buffer landed
    } else {
      asm volatile("s_wait_asynccnt 0" ::: "memory");
    }
    __syncthreads();                   // all waves' async copies visible

    // per-lane WMMA fragments (16-bit 16x32 layout: K = hi2*8.. and 16+hi2*8..)
    v16bf af[2], bfr[4];
    #pragma unroll
    for (int mi = 0; mi < 2; mi++) {
      int rr = wm + mi * 16 + l;
      v8bf lo  = *(v8bf*)(&As[buf][rr][hi2 * 8]);
      v8bf hi8 = *(v8bf*)(&As[buf][rr][16 + hi2 * 8]);
      #pragma unroll
      for (int j = 0; j < 8; j++) { af[mi][j] = lo[j]; af[mi][j + 8] = hi8[j]; }
    }
    #pragma unroll
    for (int ni = 0; ni < 4; ni++) {
      int rr = wn + ni * 16 + l;
      v8bf lo  = *(v8bf*)(&Bs[buf][rr][hi2 * 8]);
      v8bf hi8 = *(v8bf*)(&Bs[buf][rr][16 + hi2 * 8]);
      #pragma unroll
      for (int j = 0; j < 8; j++) { bfr[ni][j] = lo[j]; bfr[ni][j + 8] = hi8[j]; }
    }

    #pragma unroll
    for (int mi = 0; mi < 2; mi++)
      #pragma unroll
      for (int ni = 0; ni < 4; ni++)
        acc[mi][ni] = __builtin_amdgcn_wmma_f32_16x16x32_bf16(
            false, af[mi], false, bfr[ni], (short)0, acc[mi][ni], false, false);

    __syncthreads();                   // done reading buf before it is re-staged
    buf ^= 1;
  }

  // D layout: VGPR r -> M = r + 8*hi2 ; N = lane&15
  #pragma unroll
  for (int mi = 0; mi < 2; mi++)
    #pragma unroll
    for (int ni = 0; ni < 4; ni++)
      #pragma unroll
      for (int r = 0; r < 8; r++) {
        int m = row0 + wm + mi * 16 + hi2 * 8 + r;
        int n = col0 + wn + ni * 16 + l;
        store_c(&C[(size_t)m * ldc + n], acc[mi][ni][r] * alpha);
      }
}

// ---------------------------------------------------------------------------
// Launch
// ---------------------------------------------------------------------------
extern "C" void kernel_launch(void* const* d_in, const int* in_sizes, int n_in,
                              void* d_out, int out_size, void* d_ws, size_t ws_size,
                              hipStream_t stream) {
  const float* x        = (const float*)d_in[0];
  const float* Wq_down  = (const float*)d_in[1];
  const float* q_norm_w = (const float*)d_in[2];
  const float* Wq_up    = (const float*)d_in[3];
  const float* Wq_rope  = (const float*)d_in[4];
  const float* Wkv_down = (const float*)d_in[5];
  const float* kv_norm_w= (const float*)d_in[6];
  const float* Wk_up    = (const float*)d_in[7];
  const float* Wv_up    = (const float*)d_in[8];
  const float* Wk_rope  = (const float*)d_in[9];
  const float* Wout     = (const float*)d_in[10];

  float* out_f  = (float*)d_out;                       // [B,S,DM]
  float* attn_f = out_f + (size_t)B_ * S_ * DM_;       // [B,H,S,S]

  char* ws = (char*)d_ws;
  size_t off = 0;
  auto alloc = [&](size_t bytes) { char* p = ws + off; off += (bytes + 255) & ~(size_t)255; return p; };

  bf16_t* x_bf   = (bf16_t*)alloc((size_t)B_ * S_ * DM_ * 2);
  bf16_t* wqd_t  = (bf16_t*)alloc((size_t)DM_ * DQL_ * 2);
  bf16_t* wqu_t  = (bf16_t*)alloc((size_t)DQL_ * (H_ * HD_) * 2);
  bf16_t* wqr_t  = (bf16_t*)alloc((size_t)DM_ * (H_ * RD_) * 2);
  bf16_t* wkvd_t = (bf16_t*)alloc((size_t)DM_ * DKVL_ * 2);
  bf16_t* wku_t  = (bf16_t*)alloc((size_t)DKVL_ * (H_ * HD_) * 2);
  bf16_t* wvu_t  = (bf16_t*)alloc((size_t)DKVL_ * (H_ * HD_) * 2);
  bf16_t* wkr_t  = (bf16_t*)alloc((size_t)DM_ * (H_ * RD_) * 2);
  bf16_t* wout_t = (bf16_t*)alloc((size_t)(H_ * HD_) * DM_ * 2);
  bf16_t* cq_bf  = (bf16_t*)alloc((size_t)B_ * S_ * DQL_ * 2);
  bf16_t* ckv_bf = (bf16_t*)alloc((size_t)B_ * S_ * DKVL_ * 2);
  bf16_t* qcat   = (bf16_t*)alloc((size_t)B_ * H_ * S_ * (HD_ + RD_) * 2);
  bf16_t* kcat   = (bf16_t*)alloc((size_t)B_ * H_ * S_ * (HD_ + RD_) * 2);
  bf16_t* vt     = (bf16_t*)alloc((size_t)B_ * H_ * HD_ * S_ * 2);
  bf16_t* outh1  = (bf16_t*)alloc((size_t)B_ * H_ * S_ * HD_ * 2);   // [b,h,s,d]
  bf16_t* outh2  = (bf16_t*)alloc((size_t)B_ * H_ * S_ * HD_ * 2);   // [b,s,h*d]
  float*  cq_raw = (float*)alloc((size_t)B_ * S_ * DQL_ * 4);
  float*  ckv_raw= (float*)alloc((size_t)B_ * S_ * DKVL_ * 4);
  // TEMP region: projection raws, later aliased by bf16 attention matrix
  char*   temp   = alloc((size_t)B_ * H_ * S_ * S_ * 2);             // 64 MiB
  float*  qr_raw = (float*)(temp);
  float*  kr_raw = (float*)(temp + (size_t)B_ * S_ * (H_ * RD_) * 4);
  float*  q_raw  = (float*)(temp + (size_t)B_ * S_ * (H_ * RD_) * 8);
  float*  k_raw  = q_raw + (size_t)B_ * S_ * (H_ * HD_);
  float*  v_raw  = k_raw + (size_t)B_ * S_ * (H_ * HD_);
  bf16_t* attn_bf = (bf16_t*)temp;   // alias: raws are dead before this is written

  const int M = B_ * S_;             // 2048

  // 1) converts / transposes
  {
    int n = B_ * S_ * DM_;
    k_f32_to_bf16<<<(n + 255) / 256, 256, 0, stream>>>(x, x_bf, n);
  }
  auto T = [&](const float* w, bf16_t* wt, int K, int N) {
    int n = K * N;
    k_transpose_bf16<<<(n + 255) / 256, 256, 0, stream>>>(w, wt, K, N);
  };
  T(Wq_down,  wqd_t,  DM_,  DQL_);
  T(Wq_up,    wqu_t,  DQL_, H_ * HD_);
  T(Wq_rope,  wqr_t,  DM_,  H_ * RD_);
  T(Wkv_down, wkvd_t, DM_,  DKVL_);
  T(Wk_up,    wku_t,  DKVL_, H_ * HD_);
  T(Wv_up,    wvu_t,  DKVL_, H_ * HD_);
  T(Wk_rope,  wkr_t,  DM_,  H_ * RD_);
  T(Wout,     wout_t, H_ * HD_, DM_);

  auto gemm_f = [&](const bf16_t* A, const bf16_t* Bt, float* C, int m, int n, int k,
                    int lda, int ldb, int ldc, long long Az, long long Bz, long long Cz,
                    int batch, float alpha) {
    dim3 g(n / 128, m / 128, batch);
    k_gemm_bf16<float><<<g, 256, 0, stream>>>(A, Bt, C, m, n, k, lda, ldb, ldc, Az, Bz, Cz, alpha);
  };
  auto gemm_b = [&](const bf16_t* A, const bf16_t* Bt, bf16_t* C, int m, int n, int k,
                    int lda, int ldb, int ldc, long long Az, long long Bz, long long Cz,
                    int batch, float alpha) {
    dim3 g(n / 128, m / 128, batch);
    k_gemm_bf16<bf16_t><<<g, 256, 0, stream>>>(A, Bt, C, m, n, k, lda, ldb, ldc, Az, Bz, Cz, alpha);
  };

  // 2) down / rope projections (A = x_bf, K = DM)
  gemm_f(x_bf, wqd_t,  cq_raw,  M, DQL_,      DM_, DM_, DM_, DQL_,      0, 0, 0, 1, 1.0f);
  gemm_f(x_bf, wkvd_t, ckv_raw, M, DKVL_,     DM_, DM_, DM_, DKVL_,     0, 0, 0, 1, 1.0f);
  gemm_f(x_bf, wqr_t,  qr_raw,  M, H_ * RD_,  DM_, DM_, DM_, H_ * RD_,  0, 0, 0, 1, 1.0f);
  gemm_f(x_bf, wkr_t,  kr_raw,  M, H_ * RD_,  DM_, DM_, DM_, H_ * RD_,  0, 0, 0, 1, 1.0f);

  // 3) rmsnorm -> bf16 latents
  k_rmsnorm<<<M, 128, 0, stream>>>(cq_raw,  q_norm_w,  cq_bf,  DQL_);
  k_rmsnorm<<<M, 128, 0, stream>>>(ckv_raw, kv_norm_w, ckv_bf, DKVL_);

  // 4) up projections
  gemm_f(cq_bf,  wqu_t, q_raw, M, H_ * HD_, DQL_,  DQL_,  DQL_,  H_ * HD_, 0, 0, 0, 1, 1.0f);
  gemm_f(ckv_bf, wku_t, k_raw, M, H_ * HD_, DKVL_, DKVL_, DKVL_, H_ * HD_, 0, 0, 0, 1, 1.0f);
  gemm_f(ckv_bf, wvu_t, v_raw, M, H_ * HD_, DKVL_, DKVL_, DKVL_, H_ * HD_, 0, 0, 0, 1, 1.0f);

  // 5) rope + pack into qcat/kcat (192-wide) and v^T
  k_pack<<<B_ * H_ * S_, 128, 0, stream>>>(q_raw, k_raw, v_raw, qr_raw, kr_raw, qcat, kcat, vt);

  // 6) scores = scale * qcat @ kcat^T, batched over B*H, straight into d_out attn region
  const float scale = 1.0f / sqrtf((float)(HD_ + RD_));
  gemm_f(qcat, kcat, attn_f, S_, S_, HD_ + RD_,
         HD_ + RD_, HD_ + RD_, S_,
         (long long)S_ * (HD_ + RD_), (long long)S_ * (HD_ + RD_), (long long)S_ * S_,
         B_ * H_, scale);

  // 7) softmax in place (f32 attn output) + bf16 copy for attn@V
  k_softmax<<<B_ * H_ * S_, 256, 0, stream>>>(attn_f, attn_bf);

  // 8) out_heads = attn @ V, batched over B*H
  gemm_b(attn_bf, vt, outh1, S_, HD_, S_,
         S_, S_, HD_,
         (long long)S_ * S_, (long long)HD_ * S_, (long long)S_ * HD_,
         B_ * H_, 1.0f);

  // 9) repack heads and final projection -> d_out[0 : B*S*DM]
  {
    int n = B_ * H_ * S_ * HD_;
    k_repack_out<<<(n + 255) / 256, 256, 0, stream>>>(outh1, outh2);
  }
  gemm_f(outh2, wout_t, out_f, M, DM_, H_ * HD_,
         H_ * HD_, H_ * HD_, DM_, 0, 0, 0, 1, 1.0f);
}
